// GRU_83906481094863
// MI455X (gfx1250) — matrix operand order repaired and checked by reading, not compiled, not run on hardware
//
#include <hip/hip_runtime.h>
#include <hip/hip_bf16.h>
#include <math.h>

typedef __bf16 bf16_t;
typedef __bf16 v16bf __attribute__((ext_vector_type(16)));
typedef float  v8f   __attribute__((ext_vector_type(8)));
typedef unsigned int u32x4 __attribute__((ext_vector_type(4)));
typedef int          i32x8 __attribute__((ext_vector_type(8)));
typedef int          i32x4 __attribute__((ext_vector_type(4)));

#define T_STEPS 1024
#define D_IN    46
#define XPAD    64            // padded K for the input GEMM (2 K-tiles of 32)
#define H_DIM   128
#define B_BATCH 512
#define G3      (3 * H_DIM)   // 384 gate columns
#define NT_PER_WAVE 3         // 8 waves * 3 = 24 N-tiles = 384 cols
#define THREADS 256           // 8 wave32s

#ifndef USE_TDM
#define USE_TDM 1
#endif
#if USE_TDM && defined(__has_builtin)
#  if __has_builtin(__builtin_amdgcn_tensor_load_to_lds)
#    define HAVE_TDM 1
#  endif
#endif
#ifndef HAVE_TDM
#define HAVE_TDM 0
#endif

struct Smem {
  float  gsum[16 * 256];                       // r,z pre-activations (x+h+biases) 16 KB
  float  gxn [16 * H_DIM];                     // n-gate x-side (+b_ih)             8 KB
  float  ggn [16 * H_DIM];                     // n-gate h-side (+b_hh)             8 KB
  float  hbuf[16 * H_DIM];                     // hidden state f32                  8 KB
  __align__(16) bf16_t hbf[16 * H_DIM];        // hidden state bf16 (A-matrix src)  4 KB
  __align__(16) float  xraw[2][16 * XPAD];     // double-buffered x_t f32 tiles     8 KB
};

#if HAVE_TDM
// One TDM descriptor: copy 16 rows x 46 f32 of history (row stride T*D elems)
// into LDS at a 64-element row stride, via iterate mode (1x46 tile, 16 iters).
__device__ __forceinline__ void tdm_load_xt(const float* gptr, unsigned lds_byte_off) {
  unsigned long long ga = (unsigned long long)(uintptr_t)gptr;
  u32x4 g0;
  g0.x = 1u;                                          // count=1 (valid user D#)
  g0.y = lds_byte_off;                                // LDS dest (bytes)
  g0.z = (unsigned)ga;                                // global_addr[31:0]
  g0.w = (unsigned)((ga >> 32) & 0x1FFFFFFu) | (2u << 30); // addr[56:32] | type=2
  i32x8 g1;
  g1[0] = (int)((2u << 16) | (1u << 19));             // data_size=4B, iterate_enable
  g1[1] = (int)(46u << 16);                           // tensor_dim0 = 46
  g1[2] = (int)(1u  << 16);                           // tensor_dim1 = 1
  g1[3] = (int)(46u << 16);                           // tile_dim0   = 46
  g1[4] = 1;                                          // tile_dim1   = 1
  g1[5] = (int)(T_STEPS * D_IN);                      // tensor_dim0_stride
  g1[6] = 0;
  g1[7] = 0;
  i32x4 g2;
  g2[0] = 0;                                          // tensor_dim2
  g2[1] = XPAD;                                       // lds_addr_increment (elems)
  g2[2] = (int)(T_STEPS * D_IN);                      // global_addr_increment (elems)
  g2[3] = (int)(15u << 16);                           // iterate_count = 15 (16 rows)
  i32x4 g3 = {0, 0, 0, 0};
  i32x8 g4 = {0, 0, 0, 0, 0, 0, 0, 0};                // clang-23 extra group (unused)
  __builtin_amdgcn_tensor_load_to_lds(g0, g1, g2, g3, g4, 0);
}
#endif

__global__ __launch_bounds__(THREADS, 1)
void gru_persistent(const float* __restrict__ history,
                    const float* __restrict__ W_ih,
                    const float* __restrict__ W_hh,
                    const float* __restrict__ b_ih,
                    const float* __restrict__ b_hh,
                    const float* __restrict__ h0,
                    float* __restrict__ out)
{
  __shared__ Smem sm;
  const int tid   = threadIdx.x;
  const int lane  = tid & 31;        // wave32
  const int wave  = tid >> 5;        // 0..7
  const int nlane = lane & 15;       // column (B/C/D) / row m (A)
  const int hi    = lane >> 4;       // lane half selects K sub-run
  const int bc0   = blockIdx.x * 16; // batch tile origin

  // ---------------- weight B-fragments -> registers (bf16), once ----------
  v16bf whf[NT_PER_WAVE][4];  // W_hh: K=128 -> 4 K-tiles
  v16bf wxf[NT_PER_WAVE][2];  // W_ih: K=46 padded to 64 -> 2 K-tiles
  float bx[NT_PER_WAVE], bh[NT_PER_WAVE];

  #pragma unroll
  for (int j = 0; j < NT_PER_WAVE; ++j) {
    const int nt = wave * NT_PER_WAVE + j;
    const int n  = nt * 16 + nlane;                 // gate column 0..383
    bx[j] = b_ih[n];
    bh[j] = b_hh[n];
    #pragma unroll
    for (int kt = 0; kt < 4; ++kt) {                // B[k][n] = W_hh[n][k]
      const float* p = W_hh + (size_t)n * H_DIM + kt * 32 + hi * 8;
      v16bf f;
      #pragma unroll
      for (int i = 0; i < 16; ++i) f[i] = (bf16_t)p[i + (i & 8)];
      whf[j][kt] = f;
    }
    #pragma unroll
    for (int kt = 0; kt < 2; ++kt) {
      v16bf f;
      #pragma unroll
      for (int i = 0; i < 16; ++i) {
        const int k = kt * 32 + hi * 8 + i + (i & 8);
        f[i] = (k < D_IN) ? (bf16_t)W_ih[(size_t)n * D_IN + k] : (bf16_t)0.0f;
      }
      wxf[j][kt] = f;
    }
  }

  // ---------------- kick off DMA of x_0 while we init ----------------------
#if HAVE_TDM
  if (wave == 0)
    tdm_load_xt(history + (size_t)bc0 * T_STEPS * D_IN,
                (unsigned)(uintptr_t)&sm.xraw[0][0]);
#else
  for (int i = tid; i < 16 * D_IN; i += THREADS) {
    const int r = i / D_IN, d = i - r * D_IN;
    sm.xraw[0][r * XPAD + d] = history[((size_t)(bc0 + r) * T_STEPS) * D_IN + d];
  }
#endif

  // init h state; zero x K-padding cols 46..63 once (TDM never writes them)
  for (int i = tid; i < 16 * H_DIM; i += THREADS) {
    const float h = h0[i & (H_DIM - 1)];
    sm.hbuf[i] = h;
    sm.hbf[i]  = (bf16_t)h;
  }
  for (int i = tid; i < 2 * 16 * XPAD; i += THREADS)
    if ((i & (XPAD - 1)) >= D_IN) sm.xraw[0][i] = 0.0f;   // flat over both buffers

#if HAVE_TDM
  if (wave == 0) __builtin_amdgcn_s_wait_tensorcnt(0);
#endif
  __syncthreads();

  // ---------------- sequential scan over T --------------------------------
  for (int t = 0; t < T_STEPS; ++t) {
    const int cur = t & 1, nxt = cur ^ 1;

    // ---- A fragments (shared by this wave's 3 N-tiles) ----
    v16bf ha[4], xa[2];
    #pragma unroll
    for (int kt = 0; kt < 4; ++kt) {   // hidden: row m = nlane, runs of 8 bf16
      const int base = nlane * H_DIM + kt * 32 + hi * 8;
      union { uint4 u[2]; v16bf v; } u;
      u.u[0] = *(const uint4*)&sm.hbf[base];
      u.u[1] = *(const uint4*)&sm.hbf[base + 16];
      ha[kt] = u.v;
    }
    #pragma unroll
    for (int kt = 0; kt < 2; ++kt) {   // input: f32 rows at stride 64, cvt->bf16
      const float* q = &sm.xraw[cur][nlane * XPAD + kt * 32 + hi * 8];
      const float4 a0 = *(const float4*)(q +  0);
      const float4 a1 = *(const float4*)(q +  4);
      const float4 b0 = *(const float4*)(q + 16);
      const float4 b1 = *(const float4*)(q + 20);
      v16bf f;
      f[0]=(bf16_t)a0.x; f[1]=(bf16_t)a0.y; f[2]=(bf16_t)a0.z; f[3]=(bf16_t)a0.w;
      f[4]=(bf16_t)a1.x; f[5]=(bf16_t)a1.y; f[6]=(bf16_t)a1.z; f[7]=(bf16_t)a1.w;
      f[8]=(bf16_t)b0.x; f[9]=(bf16_t)b0.y; f[10]=(bf16_t)b0.z; f[11]=(bf16_t)b0.w;
      f[12]=(bf16_t)b1.x; f[13]=(bf16_t)b1.y; f[14]=(bf16_t)b1.z; f[15]=(bf16_t)b1.w;
      xa[kt] = f;
    }

    // ---- 3 N-tiles: dual accumulators (n-gate needs x/h parts split) ----
    #pragma unroll
    for (int j = 0; j < NT_PER_WAVE; ++j) {
      v8f accx = {}; v8f acch = {};
      #pragma unroll
      for (int kt = 0; kt < 2; ++kt)
        accx = __builtin_amdgcn_wmma_f32_16x16x32_bf16(
                 false, xa[kt], false, wxf[j][kt], (short)0, accx, false, false);
      #pragma unroll
      for (int kt = 0; kt < 4; ++kt)
        acch = __builtin_amdgcn_wmma_f32_16x16x32_bf16(
                 false, ha[kt], false, whf[j][kt], (short)0, acch, false, false);
      const int nt  = wave * NT_PER_WAVE + j;
      const int col = nt * 16 + nlane;
      if (nt < 16) {                    // r/z gates: pre-sum x+h (+ both biases)
        #pragma unroll
        for (int v = 0; v < 8; ++v)
          sm.gsum[(v + hi * 8) * 256 + col] = accx[v] + acch[v] + bx[j] + bh[j];
      } else {                          // n gate: keep parts separate
        const int c2 = col - 256;
        #pragma unroll
        for (int v = 0; v < 8; ++v) {
          sm.gxn[(v + hi * 8) * H_DIM + c2] = accx[v] + bx[j];
          sm.ggn[(v + hi * 8) * H_DIM + c2] = acch[v] + bh[j];
        }
      }
    }

    // ---- prefetch x_{t+1} via Tensor Data Mover (overlaps gate phase) ----
    if (t + 1 < T_STEPS) {
#if HAVE_TDM
      if (wave == 0)
        tdm_load_xt(history + ((size_t)bc0 * T_STEPS + (t + 1)) * D_IN,
                    (unsigned)(uintptr_t)&sm.xraw[nxt][0]);
#else
      for (int i = tid; i < 16 * D_IN; i += THREADS) {
        const int r = i / D_IN, d = i - r * D_IN;
        sm.xraw[nxt][r * XPAD + d] =
            history[((size_t)(bc0 + r) * T_STEPS + (t + 1)) * D_IN + d];
      }
#endif
    }
    __syncthreads();

    // ---- gate combine: thread -> row m, 8 consecutive hidden columns ----
    {
      const int m  = tid >> 4;
      const int j0 = (tid & 15) * 8;
      #pragma unroll
      for (int q = 0; q < 8; ++q) {
        const int jj = j0 + q;
        const float r = 1.0f / (1.0f + __expf(-sm.gsum[m * 256 + jj]));
        const float z = 1.0f / (1.0f + __expf(-sm.gsum[m * 256 + 128 + jj]));
        const float nn = tanhf(sm.gxn[m * H_DIM + jj] + r * sm.ggn[m * H_DIM + jj]);
        const float hprev = sm.hbuf[m * H_DIM + jj];
        const float hnew  = (1.0f - z) * nn + z * hprev;
        sm.hbuf[m * H_DIM + jj] = hnew;
        sm.hbf[m * H_DIM + jj]  = (bf16_t)hnew;
      }
    }
#if HAVE_TDM
    if (wave == 0) __builtin_amdgcn_s_wait_tensorcnt(0);  // x_{t+1} landed
#endif
    __syncthreads();
  }

  // ---------------- write final hidden state ------------------------------
  for (int i = tid; i < 16 * H_DIM; i += THREADS) {
    const int m = i >> 7, k = i & 127;
    out[(size_t)(bc0 + m) * H_DIM + k] = sm.hbuf[i];
  }
}

extern "C" void kernel_launch(void* const* d_in, const int* in_sizes, int n_in,
                              void* d_out, int out_size, void* d_ws, size_t ws_size,
                              hipStream_t stream) {
  const float* history = (const float*)d_in[0];
  const float* W_ih    = (const float*)d_in[1];
  const float* W_hh    = (const float*)d_in[2];
  const float* b_ih    = (const float*)d_in[3];
  const float* b_hh    = (const float*)d_in[4];
  const float* h0      = (const float*)d_in[5];
  (void)in_sizes; (void)n_in; (void)out_size; (void)d_ws; (void)ws_size;

  dim3 grid(B_BATCH / 16);   // 32 independent batch-tile recurrences
  gru_persistent<<<grid, THREADS, 0, stream>>>(
      history, W_ih, W_hh, b_ih, b_hh, h0, (float*)d_out);
}
